// MultiHeadSelfAttention_30889404792993
// MI455X (gfx1250) — matrix operand order repaired and checked by reading, not compiled, not run on hardware
//
#include <hip/hip_runtime.h>
#include <hip/hip_bf16.h>

// Problem constants (reference: B=2, M=2048, HS=1024, K=16 heads, D=64)
#define BDIM 2
#define MDIM 2048
#define HSZ  1024
#define KH   16
#define DH   64
#define NHEAD (BDIM*KH)
#define NROWS (BDIM*MDIM)   // 4096 rows of h

typedef __attribute__((ext_vector_type(16))) __bf16          v16bf;
typedef __attribute__((ext_vector_type(8)))  float           v8f;
typedef __attribute__((ext_vector_type(8)))  unsigned short  v8u16;

union ABf {
    v16bf          bf;
    v8u16          h2[2];
    unsigned short u[16];
};

__device__ __forceinline__ unsigned short f2bf(float f) {
    unsigned int u = __float_as_uint(f);
    u += 0x7fffu + ((u >> 16) & 1u);     // round to nearest even
    return (unsigned short)(u >> 16);
}

// ---------------------------------------------------------------- convert f32 -> bf16
__global__ void __launch_bounds__(256) cvt_f32_bf16(const float* __restrict__ in,
                                                    unsigned short* __restrict__ out, int n) {
    int i = blockIdx.x * blockDim.x + threadIdx.x;
    if (i < n) out[i] = f2bf(in[i]);
}

// ---------------------------------------------------------------- span projection (f32 VALU, precision-sensitive)
// mean/soft layout: [head = b*KH+kh][m], head-major
__global__ void __launch_bounds__(256) span_kernel(const float* __restrict__ h,
                                                   const float* __restrict__ Wspan,
                                                   float* __restrict__ mean,
                                                   float* __restrict__ soft) {
    int idx = blockIdx.x * blockDim.x + threadIdx.x;   // (b*KH+kh)*MDIM + m
    if (idx >= NHEAD * MDIM) return;
    int m    = idx % MDIM;
    int head = idx / MDIM;
    int b    = head / KH;
    int kh   = head % KH;
    const float* hr = h + ((size_t)b * MDIM + m) * HSZ;
    const float* w0 = Wspan + (size_t)(2 * kh) * HSZ;      // span[...,0]
    const float* w1 = w0 + HSZ;                            // span[...,1]
    float s0 = 0.f, s1 = 0.f;
    for (int k = 0; k < HSZ; k += 4) {
        float4 hv = *(const float4*)(hr + k);
        float4 a  = *(const float4*)(w0 + k);
        float4 c  = *(const float4*)(w1 + k);
        s0 += hv.x * a.x + hv.y * a.y + hv.z * a.z + hv.w * a.w;
        s1 += hv.x * c.x + hv.y * c.y + hv.z * c.z + hv.w * c.w;
    }
    float mu = (float)MDIM / (1.0f + __expf(-s0));                     // sigmoid * M
    float sp = (s1 > 20.0f) ? s1 : __logf(1.0f + __expf(s1));          // softplus
    mean[idx] = mu;
    soft[idx] = sp;
}

// ---------------------------------------------------------------- WMMA tile loads
// A tile (16x32 bf16, row-major source, 16-bit A layout):
//   lane L (r = L&15, g = L>>4): elem e -> K = (e<8 ? e : e+8) + 8*g
__device__ __forceinline__ void load_a_tile(const unsigned short* __restrict__ A, int lda,
                                            int lane, int kbase, ABf& a) {
    int g = lane >> 4;
    int r = lane & 15;
    const unsigned short* p = A + (size_t)r * lda + kbase + 8 * g;
    a.h2[0] = *(const v8u16*)p;
    a.h2[1] = *(const v8u16*)(p + 16);
}

// B tile (32x16 bf16) built from W rows (B[k,n] = W[n,k]):
//   lane L (n = L&15, g = L>>4): elem e -> K = (e<8 ? e : e+8) + 8*g, along W row n
__device__ __forceinline__ void load_b_tile(const unsigned short* __restrict__ W, int ldb,
                                            int lane, int kbase, ABf& b) {
    int g = lane >> 4;
    int n = lane & 15;
    const unsigned short* p = W + (size_t)n * ldb + kbase + 8 * g;
    b.h2[0] = *(const v8u16*)p;
    b.h2[1] = *(const v8u16*)(p + 16);
}

// 32(M) x 64(N) x HSZ(K) MMA core: 2 A-tiles x 4 B-tiles, B reused across A rows
__device__ __forceinline__ void mma_32x64(const unsigned short* __restrict__ A0,
                                          const unsigned short* __restrict__ A1,
                                          const unsigned short* __restrict__ Wb,  // offset to ng*64 rows
                                          int lane, v8f acc[2][4]) {
    for (int kb = 0; kb < HSZ; kb += 32) {
        __builtin_prefetch(A0 + ((lane & 15) * HSZ) + kb + 64, 0, 1);
        ABf a0, a1;
        load_a_tile(A0, HSZ, lane, kb, a0);
        load_a_tile(A1, HSZ, lane, kb, a1);
#pragma unroll
        for (int t = 0; t < 4; ++t) {
            ABf b; load_b_tile(Wb + (size_t)(t * 16) * HSZ, HSZ, lane, kb, b);
            acc[0][t] = __builtin_amdgcn_wmma_f32_16x16x32_bf16(
                false, a0.bf, false, b.bf, (short)0, acc[0][t], false, false);
            acc[1][t] = __builtin_amdgcn_wmma_f32_16x16x32_bf16(
                false, a1.bf, false, b.bf, (short)0, acc[1][t], false, false);
        }
    }
}

// ---------------------------------------------------------------- value = h @ W_val.T, stored head-transposed bf16
// vt layout: [(b*KH+kh)*DH + d][m]  (M contiguous) -> contiguous B-tiles for attention
__global__ void __launch_bounds__(256) gemm_value(const unsigned short* __restrict__ hbf,
                                                  const unsigned short* __restrict__ wbf,
                                                  unsigned short* __restrict__ vt) {
    int wave = (int)((blockIdx.x * blockDim.x + threadIdx.x) >> 5);
    int lane = threadIdx.x & 31;
    const int MT = NROWS / 32;      // 128 m-tiles of 32 rows
    int mt = wave % MT;
    int ng = wave / MT;             // 16 n-groups of 64 columns
    if (ng >= HSZ / 64) return;

    const unsigned short* A0 = hbf + (size_t)mt * 32 * HSZ;
    const unsigned short* A1 = A0 + (size_t)16 * HSZ;
    v8f acc[2][4] = {};
    mma_32x64(A0, A1, wbf + (size_t)ng * 64 * HSZ, lane, acc);

    // D layout: VGPR r, lane l -> row = r + 8*(l>>4), col fixed per lane -> packed b128 store
    int g = lane >> 4;
#pragma unroll
    for (int s = 0; s < 2; ++s) {
        int mglob = mt * 32 + s * 16;
        int b  = mglob / MDIM;
        int m0 = mglob % MDIM;
#pragma unroll
        for (int t = 0; t < 4; ++t) {
            int n  = ng * 64 + t * 16 + (lane & 15);   // global column in HS
            int kh = n / DH, d = n % DH;
            v8u16 pk;
#pragma unroll
            for (int r = 0; r < 8; ++r) pk[r] = f2bf(acc[s][t][r]);
            size_t off = ((size_t)(b * KH + kh) * DH + d) * MDIM + m0 + 8 * g;
            *(v8u16*)(vt + off) = pk;
        }
    }
}

// ---------------------------------------------------------------- streaming Gaussian attention (flash-style, closed-form rowmax)
// one wave per (head, 16-row i-tile); output bf16 into (B, M, HS) layout
__global__ void __launch_bounds__(256) attn_kernel(const float* __restrict__ mean,
                                                   const float* __restrict__ soft,
                                                   const unsigned short* __restrict__ vt,
                                                   unsigned short* __restrict__ aout) {
    int wave = (int)((blockIdx.x * blockDim.x + threadIdx.x) >> 5);
    int lane = threadIdx.x & 31;
    const int IT = MDIM / 16;       // 128 i-tiles per head
    int head = wave / IT;
    if (head >= NHEAD) return;
    int it = wave % IT;
    int b  = head / KH;
    int kh = head % KH;
    int i0 = it * 16;
    int r  = lane & 15;
    int g  = lane >> 4;

    float mu = mean[(size_t)head * MDIM + i0 + r];
    float sf = soft[(size_t)head * MDIM + i0 + r];
    // closed-form row max: attained at nearest in-range integer position
    float js   = fminf(fmaxf(roundf(mu), 0.0f), (float)(MDIM - 1));
    float dmx  = js - mu;
    float rmax = -sf * dmx * dmx;

    v8f acc[4] = {};
    float psum = 0.0f;
    const unsigned short* Vh = vt + (size_t)head * DH * MDIM;

    for (int jb = 0; jb < MDIM; jb += 32) {
        ABf a;
#pragma unroll
        for (int e = 0; e < 16; ++e) {
            int kk  = (e < 8 ? e : e + 8) + 8 * g;          // 16-bit A-layout K mapping
            float j = (float)(jb + kk);
            float d = j - mu;
            float p = __expf(-sf * d * d - rmax);           // <= 1, no rescale needed
            psum += p;
            a.u[e] = f2bf(p);
        }
#pragma unroll
        for (int t = 0; t < 4; ++t) {
            // B tile: value rows j=jb..jb+31, cols d = t*16..+15; contiguous in vt
            const unsigned short* p = Vh + (size_t)(t * 16 + (lane & 15)) * MDIM + jb + 8 * g;
            ABf bm;
            bm.h2[0] = *(const v8u16*)p;
            bm.h2[1] = *(const v8u16*)(p + 16);
            acc[t] = __builtin_amdgcn_wmma_f32_16x16x32_bf16(
                false, a.bf, false, bm.bf, (short)0, acc[t], false, false);
        }
    }
    // row r is split across lanes r and r+16 (disjoint j subsets) -> combine
    psum += __shfl_xor(psum, 16, 32);

#pragma unroll
    for (int t = 0; t < 4; ++t) {
#pragma unroll
        for (int rr = 0; rr < 8; ++rr) {
            int row     = rr + 8 * g;                        // D-layout row
            float denom = __shfl(psum, row, 32);             // >= 1 always
            float val   = acc[t][rr] / denom;
            int mrow = i0 + row;
            int col  = kh * DH + t * 16 + (lane & 15);       // head-interleaved back to HS
            aout[((size_t)b * MDIM + mrow) * HSZ + col] = f2bf(val);
        }
    }
}

// ---------------------------------------------------------------- final: out = attn_out @ W_out.T  (f32 result)
__global__ void __launch_bounds__(256) gemm_out(const unsigned short* __restrict__ abf,
                                                const unsigned short* __restrict__ wbf,
                                                float* __restrict__ out) {
    int wave = (int)((blockIdx.x * blockDim.x + threadIdx.x) >> 5);
    int lane = threadIdx.x & 31;
    const int MT = NROWS / 32;
    int mt = wave % MT;
    int ng = wave / MT;
    if (ng >= HSZ / 64) return;

    const unsigned short* A0 = abf + (size_t)mt * 32 * HSZ;
    const unsigned short* A1 = A0 + (size_t)16 * HSZ;
    v8f acc[2][4] = {};
    mma_32x64(A0, A1, wbf + (size_t)ng * 64 * HSZ, lane, acc);

    int g = lane >> 4;
#pragma unroll
    for (int s = 0; s < 2; ++s) {
#pragma unroll
        for (int t = 0; t < 4; ++t) {
            int n = ng * 64 + t * 16 + (lane & 15);
#pragma unroll
            for (int r = 0; r < 8; ++r) {
                out[(size_t)(mt * 32 + s * 16 + 8 * g + r) * HSZ + n] = acc[s][t][r];
            }
        }
    }
}

// ---------------------------------------------------------------- host launcher
extern "C" void kernel_launch(void* const* d_in, const int* in_sizes, int n_in,
                              void* d_out, int out_size, void* d_ws, size_t ws_size,
                              hipStream_t stream) {
    const float* h     = (const float*)d_in[0];   // (B, M, HS)
    const float* Wspan = (const float*)d_in[1];   // (2K, HS)
    const float* Wval  = (const float*)d_in[2];   // (HS, HS)
    const float* Wout  = (const float*)d_in[3];   // (HS, HS)
    float* out = (float*)d_out;                   // (B, M, HS) f32

    char* ws = (char*)d_ws;
    const size_t N_H  = (size_t)NROWS * HSZ;      // 4,194,304
    const size_t N_W  = (size_t)HSZ * HSZ;        // 1,048,576
    const size_t N_MS = (size_t)NHEAD * MDIM;     //    65,536

    unsigned short* hbf    = (unsigned short*)(ws);                            // 8 MB
    unsigned short* wvalbf = (unsigned short*)(ws + N_H * 2);                  // 2 MB
    unsigned short* woutbf = (unsigned short*)(ws + N_H * 2 + N_W * 2);        // 2 MB
    unsigned short* vt     = (unsigned short*)(ws + N_H * 2 + N_W * 4);        // 8 MB (B,K,D,M) bf16
    float*          meanp  = (float*)(ws + N_H * 4 + N_W * 4);                 // 256 KB
    float*          softp  = (float*)(ws + N_H * 4 + N_W * 4 + N_MS * 4);      // 256 KB
    unsigned short* aoutbf = (unsigned short*)(ws + N_H * 4 + N_W * 4 + N_MS * 8); // 8 MB

    // 1) convert inputs to bf16
    cvt_f32_bf16<<<(int)((N_H + 255) / 256), 256, 0, stream>>>(h, hbf, (int)N_H);
    cvt_f32_bf16<<<(int)((N_W + 255) / 256), 256, 0, stream>>>(Wval, wvalbf, (int)N_W);
    cvt_f32_bf16<<<(int)((N_W + 255) / 256), 256, 0, stream>>>(Wout, woutbf, (int)N_W);

    // 2) span projection -> mean/soft (f32 for precision)
    span_kernel<<<(int)((N_MS + 255) / 256), 256, 0, stream>>>(h, Wspan, meanp, softp);

    // 3) value projection (WMMA bf16, 32x64 tiles), stored head-transposed
    {
        int waves = (NROWS / 32) * (HSZ / 64);    // 2048 waves
        gemm_value<<<waves * 32 / 256, 256, 0, stream>>>(hbf, wvalbf, vt);
    }

    // 4) streaming Gaussian attention (WMMA bf16, closed-form rowmax softmax)
    {
        int waves = NHEAD * (MDIM / 16);          // 4096 waves
        attn_kernel<<<waves * 32 / 256, 256, 0, stream>>>(meanp, softp, vt, aoutbf);
    }

    // 5) output projection (WMMA bf16 -> f32, 32x64 tiles)
    {
        int waves = (NROWS / 32) * (HSZ / 64);    // 2048 waves
        gemm_out<<<waves * 32 / 256, 256, 0, stream>>>(aoutbf, woutbf, out);
    }
}
